// NormLayer_61881888801194
// MI455X (gfx1250) — matrix operand order repaired and checked by reading, not compiled, not run on hardware
//
#include <hip/hip_runtime.h>
#include <math.h>

typedef __attribute__((ext_vector_type(16))) _Float16 v16h;
typedef __attribute__((ext_vector_type(8)))  _Float16 v8h;
typedef __attribute__((ext_vector_type(8)))  float    v8f;

#define KN 16384
#define KD 128
#define SKP 136   // K tile row pitch in f16 (272 B: 16B-aligned, bank-conflict-free)
#define SVP 40    // V tile row pitch in f16 (80 B: 16B-aligned, bank-conflict-free)

__device__ __forceinline__ v8f wmma_f16(v16h a, v16h b, v8f c) {
    return __builtin_amdgcn_wmma_f32_16x16x32_f16(
        /*neg_a=*/false, a, /*neg_b=*/false, b,
        /*c_mod=*/(short)0, c, /*reuse_a=*/false, /*reuse_b=*/false);
}

// A-operand (16x32 f16, M x K): lane l -> row = l%16, g = l/16.
// element e<8 : K = 32c + 8g + e ; element e>=8 : K = 32c + 16 + 8g + (e-8).
__device__ __forceinline__ v16h ld_A16(const _Float16* row, int c, int g) {
    const v8h lo = *(const v8h*)(row + 32 * c + 8 * g);
    const v8h hi = *(const v8h*)(row + 32 * c + 16 + 8 * g);
    v16h r;
#pragma unroll
    for (int i = 0; i < 8; ++i) { r[i] = lo[i]; r[i + 8] = hi[i]; }
    return r;
}

// B-operand (32x16 f16, K x N): lane l -> col = l%16, g = l/16.
// element e<8 : K = base + 8g + e ; element e>=8 : K = base + 16 + 8g + (e-8)  (off16 = base + 8g).
__device__ __forceinline__ v16h ld_B16(const _Float16* row, int off16) {
    const v8h lo = *(const v8h*)(row + off16);
    const v8h hi = *(const v8h*)(row + off16 + 16);
    v16h r;
#pragma unroll
    for (int i = 0; i < 8; ++i) { r[i] = lo[i]; r[i + 8] = hi[i]; }
    return r;
}

// Async copy of 16 bytes/lane from global to LDS (CDNA5 ASYNCcnt path).
__device__ __forceinline__ void async_b128(const _Float16* g, const _Float16* l) {
    const unsigned lds_off = (unsigned)(uintptr_t)l;   // low 32 bits = LDS byte address
    asm volatile("global_load_async_to_lds_b128 %0, %1, off"
                 :: "v"(lds_off), "v"(g) : "memory");
}

// ---------------------------------------------------------------------------
// Kernel 1: per-row L2 norm; emit norm_x (f16 row-major) and x^T (f16).
// ---------------------------------------------------------------------------
__global__ __launch_bounds__(128) void prep_kernel(const float* __restrict__ x,
                                                   _Float16* __restrict__ qh,
                                                   _Float16* __restrict__ xt) {
    const int row = blockIdx.x;
    const int d   = threadIdx.x;           // 0..127
    const float v = x[(size_t)row * KD + d];

    float ss = v * v;
#pragma unroll
    for (int m = 1; m < 32; m <<= 1) ss += __shfl_xor(ss, m, 32);
    __shared__ float red[4];
    if ((threadIdx.x & 31) == 0) red[threadIdx.x >> 5] = ss;
    __syncthreads();
    const float tot  = red[0] + red[1] + red[2] + red[3];
    const float norm = sqrtf(tot);
    const float inv  = 1.0f / fmaxf(norm, 1e-12f);

    qh[(size_t)row * KD + d] = (_Float16)(v * inv);
    xt[(size_t)d * KN + row] = (_Float16)v;
}

// ---------------------------------------------------------------------------
// Kernel 2: fused flash-attention + residual-combine + LayerNorm.
// 8 waves/block, one 16-query tile per wave; K/V tiles staged in LDS via
// async global->LDS DMA, double-buffered.
// ---------------------------------------------------------------------------
__global__ __launch_bounds__(256) void flash_kernel(const float* __restrict__ x,
                                                    const float* __restrict__ gamma,
                                                    const float* __restrict__ beta,
                                                    const _Float16* __restrict__ qh,
                                                    const _Float16* __restrict__ xt,
                                                    float* __restrict__ out) {
    __shared__ _Float16 sK[2][32 * SKP];    // keys: 32 rows x 128 f16 (padded)
    __shared__ _Float16 sV[2][128 * SVP];   // V^T slice: 128 rows x 32 f16 (padded)

    const int tid   = threadIdx.x;
    const int lane  = tid & 31;
    const int wave  = tid >> 5;
    const int tile  = blockIdx.x * 8 + wave;   // 0..1023
    const int qbase = tile * 16;
    const int q     = lane & 15;
    const int g     = lane >> 4;

    // Cooperative async stage of one 32-key block into LDS buffer `b`.
    // Each thread issues 4 async b128 ops (2 for K, 2 for V) -> 4 per wave... per lane,
    // i.e. every wave's own ASYNCcnt rises by 4.
    auto load_tiles = [&](int b, int kb) {
        {   // K: 32 contiguous rows of 256B from qh + kb*KD
            const int j1 = tid + 256;
            const int r0 = tid >> 4, s0 = tid & 15;
            const int r1 = j1 >> 4,  s1 = j1 & 15;
            async_b128(qh + (size_t)(kb + r0) * KD + s0 * 8, &sK[b][r0 * SKP + s0 * 8]);
            async_b128(qh + (size_t)(kb + r1) * KD + s1 * 8, &sK[b][r1 * SKP + s1 * 8]);
        }
        {   // V: 128 rows of 64B from xt (row stride KN), columns [kb, kb+32)
            const int j1 = tid + 256;
            const int d0 = tid >> 2, c0 = tid & 3;
            const int d1 = j1 >> 2,  c1 = j1 & 3;
            async_b128(xt + (size_t)d0 * KN + kb + c0 * 8, &sV[b][d0 * SVP + c0 * 8]);
            async_b128(xt + (size_t)d1 * KN + kb + c1 * 8, &sV[b][d1 * SVP + c1 * 8]);
        }
    };

    // Preload Q as B-operand chunks (Q^T), contraction over D in 4 chunks of 32.
    v16h Qb[4];
    {
        const _Float16* qrow = qh + (size_t)(qbase + q) * KD;
#pragma unroll
        for (int c = 0; c < 4; ++c) Qb[c] = ld_B16(qrow, 32 * c + 8 * g);
    }

    v8f O[8];
#pragma unroll
    for (int t = 0; t < 8; ++t)
#pragma unroll
        for (int i = 0; i < 8; ++i) O[t][i] = 0.0f;

    float m_run = -INFINITY;
    float l_run = 0.0f;
    const v8f zero = O[0];

    load_tiles(0, 0);
    int buf = 0;

    for (int kb = 0; kb < KN; kb += 32) {
        if (kb + 32 < KN) {
            load_tiles(buf ^ 1, kb + 32);
            asm volatile("s_wait_asynccnt 4" ::: "memory");   // current buffer's 4 done
        } else {
            asm volatile("s_wait_asynccnt 0" ::: "memory");
        }
        __syncthreads();   // all waves' portions of current buffer visible

        // ---- S^T = K_block x Q^T (M = key, N = query), two 16-key tiles ----
        const _Float16* krow0 = &sK[buf][q * SKP];
        const _Float16* krow1 = &sK[buf][(16 + q) * SKP];
        v8f s0 = zero, s1 = zero;
#pragma unroll
        for (int c = 0; c < 4; ++c) {
            const v16h ka0 = ld_A16(krow0, c, g);
            const v16h ka1 = ld_A16(krow1, c, g);
            s0 = wmma_f16(ka0, Qb[c], s0);
            s1 = wmma_f16(ka1, Qb[c], s1);
        }
        // Lane l holds, for query q = l%16:
        //   s0[r] = sim(key kb+8g+r, q), s1[r] = sim(key kb+16+8g+r, q).

        // ---- online softmax (TAU = 1.0) ----
        float bmax = -INFINITY;
#pragma unroll
        for (int i = 0; i < 8; ++i) bmax = fmaxf(bmax, fmaxf(s0[i], s1[i]));
        bmax = fmaxf(bmax, __shfl_xor(bmax, 16, 32));
        const float m_new = fmaxf(m_run, bmax);

        float p0[8], p1[8], bsum = 0.0f;
#pragma unroll
        for (int i = 0; i < 8; ++i) {
            p0[i] = __expf(s0[i] - m_new);
            p1[i] = __expf(s1[i] - m_new);
            bsum += p0[i] + p1[i];
        }
        bsum += __shfl_xor(bsum, 16, 32);

        const float alpha = __expf(m_run - m_new);   // 0 on first iter
        l_run = l_run * alpha + bsum;
        m_run = m_new;

        // P lands directly in the A-operand layout (M = query, K = key rel.).
        v16h Pa;
#pragma unroll
        for (int i = 0; i < 8; ++i) {
            Pa[i]     = (_Float16)p0[i];
            Pa[i + 8] = (_Float16)p1[i];
        }

        // ---- O = alpha*O + P x V over the 8 output-column tiles ----
#pragma unroll
        for (int t = 0; t < 8; ++t) {
            const _Float16* vrow = &sV[buf][(16 * t + q) * SVP];
            const v16h Vb = ld_B16(vrow, 8 * g);
            v8f o = O[t];
#pragma unroll
            for (int i = 0; i < 8; ++i) o[i] *= alpha;
            O[t] = wmma_f16(Pa, Vb, o);
        }

        __syncthreads();   // everyone done reading buf before it is refilled
        buf ^= 1;
    }

    // ---- epilogue: x_neg = O/l ; y = 1.5x - 0.5 x_neg ; LayerNorm over D ----
    // O tile layout: lane l, VGPR r -> row (query) = 8g + r, col = 16t + q.
    float lrec[8];
#pragma unroll
    for (int r = 0; r < 8; ++r) lrec[r] = 1.0f / __shfl(l_run, 8 * g + r, 32);

    float gma[8], bta[8];
#pragma unroll
    for (int t = 0; t < 8; ++t) {
        gma[t] = gamma[16 * t + q];
        bta[t] = beta[16 * t + q];
    }

#pragma unroll
    for (int r = 0; r < 8; ++r) {
        const int row = qbase + 8 * g + r;
        const float* xr = x + (size_t)row * KD;
        float yv[8], s = 0.0f, s2 = 0.0f;
#pragma unroll
        for (int t = 0; t < 8; ++t) {
            const float xneg = O[t][r] * lrec[r];
            const float v = 1.5f * xr[16 * t + q] - 0.5f * xneg;
            yv[t] = v;
            s  += v;
            s2 += v * v;
        }
#pragma unroll
        for (int m = 1; m < 16; m <<= 1) {
            s  += __shfl_xor(s,  m, 32);
            s2 += __shfl_xor(s2, m, 32);
        }
        const float mu   = s * (1.0f / (float)KD);
        const float var  = s2 * (1.0f / (float)KD) - mu * mu;
        const float rstd = rsqrtf(var + 1e-5f);
        float* orow = out + (size_t)row * KD;
#pragma unroll
        for (int t = 0; t < 8; ++t)
            orow[16 * t + q] = (yv[t] - mu) * rstd * gma[t] + bta[t];
    }
}

extern "C" void kernel_launch(void* const* d_in, const int* in_sizes, int n_in,
                              void* d_out, int out_size, void* d_ws, size_t ws_size,
                              hipStream_t stream) {
    (void)in_sizes; (void)n_in; (void)out_size; (void)ws_size;
    const float* x     = (const float*)d_in[0];
    const float* gamma = (const float*)d_in[1];
    const float* beta  = (const float*)d_in[2];
    float* out = (float*)d_out;

    _Float16* qh = (_Float16*)d_ws;                    // norm_x, f16, [N][D]  (4 MB)
    _Float16* xt = qh + (size_t)KN * KD;               // x^T,    f16, [D][N]  (4 MB)

    prep_kernel<<<KN, 128, 0, stream>>>(x, qh, xt);
    flash_kernel<<<KN / 16 / 8, 256, 0, stream>>>(x, gamma, beta, qh, xt, out);
}
